// SeqSeqModel_72258529788539
// MI455X (gfx1250) — compile-verified, matrix operand order checked by reading, not run on hardware
//
#include <hip/hip_runtime.h>

typedef __attribute__((ext_vector_type(16))) __bf16 v16bf;
typedef __attribute__((ext_vector_type(8)))  __bf16 bf16x8;
typedef __attribute__((ext_vector_type(2)))  __bf16 bf16x2;
typedef __attribute__((ext_vector_type(8)))  float  v8f;

#define WMMA_BF16(a, b, c) \
  __builtin_amdgcn_wmma_f32_16x16x32_bf16(false, (a), false, (b), (short)0, (c), false, false)

constexpr int Bc = 8, Tc = 64, Nc = 2048, Hc = 64, Fc = 64, Ec = 32768, Gc = 192;
constexpr int BN  = Bc * Nc;        // 16384 GEMM rows
constexpr int BNH = BN * Hc;        // 1,048,576 elements per activation buffer

// ---- fragment loaders (ISA 16-bit A/B layout: lane L -> row/col L%16,
// K chunks [kb,kb+8) and [kb+16,kb+24) with kb = (L/16)*8) ------------------
__device__ inline v16bf frag_load(const __bf16* p) {
  bf16x8 lo = *(const bf16x8*)(p);
  bf16x8 hi = *(const bf16x8*)(p + 16);
  v16bf f;
#pragma unroll
  for (int i = 0; i < 8; ++i) { f[i] = lo[i]; f[i + 8] = hi[i]; }
  return f;
}
__device__ inline v16bf frag_load(const float* p) {
  v16bf f;
#pragma unroll
  for (int i = 0; i < 8; ++i) { f[i] = (__bf16)p[i]; f[i + 8] = (__bf16)p[16 + i]; }
  return f;
}
__device__ inline v16bf frag_zero() {
  v16bf f;
#pragma unroll
  for (int i = 0; i < 16; ++i) f[i] = (__bf16)0.0f;
  return f;
}

// ---------------------------------------------------------------------------
// m = state @ W_msg^T + b_msg     [B*N,64]x[64,64], all-bf16 operands
// ---------------------------------------------------------------------------
__global__ __launch_bounds__(128) void msg_kernel(const __bf16* __restrict__ state_bf,
                                                  const __bf16* __restrict__ WmsgB,
                                                  const float* __restrict__ bmsg,
                                                  __bf16* __restrict__ m_bf) {
  const int lane = threadIdx.x & 31;
  const int wave = threadIdx.x >> 5;
  const int row0 = blockIdx.x * 16;
  const int g0   = wave * 16;
  const int mrow = lane & 15;
  const int kb   = (lane >> 4) * 8;
  const int col  = g0 + (lane & 15);

  v8f acc;
  const float bias = bmsg[col];
#pragma unroll
  for (int v = 0; v < 8; ++v) acc[v] = bias;

#pragma unroll
  for (int k0 = 0; k0 < Hc; k0 += 32) {
    v16bf a = frag_load(state_bf + (size_t)(row0 + mrow) * Hc + k0 + kb);
    v16bf b = frag_load(WmsgB + (size_t)col * Hc + k0 + kb);
    acc = WMMA_BF16(a, b, acc);
  }

  const int rbase = row0 + 8 * (lane >> 4);
#pragma unroll
  for (int v = 0; v < 8; ++v)
    m_bf[(size_t)(rbase + v) * Hc + col] = (__bf16)acc[v];
}

// ---------------------------------------------------------------------------
// CSR-by-destination build (edges are launch-invariant; built once per call)
// ---------------------------------------------------------------------------
__global__ __launch_bounds__(256) void count_kernel(const int* __restrict__ edges,
                                                    int* __restrict__ counts) {
  const int e = blockIdx.x * 256 + threadIdx.x;
  if (e < Ec) atomicAdd(&counts[edges[Ec + e]], 1);
}

__global__ void scan_kernel(int* __restrict__ counts, int* __restrict__ rowptr) {
  if (blockIdx.x == 0 && threadIdx.x == 0) {
    int acc = 0;
    for (int n = 0; n < Nc; ++n) {
      const int c = counts[n];
      rowptr[n] = acc;
      acc += c;
      counts[n] = 0;                     // reuse as fill cursor
    }
    rowptr[Nc] = acc;
  }
}

__global__ __launch_bounds__(256) void fill_kernel(const int* __restrict__ edges,
                                                   const int* __restrict__ rowptr,
                                                   int* __restrict__ cursor,
                                                   int* __restrict__ srcsort) {
  const int e = blockIdx.x * 256 + threadIdx.x;
  if (e >= Ec) return;
  const int s = edges[e];
  const int d = edges[Ec + e];
  const int pos = atomicAdd(&cursor[d], 1);
  srcsort[rowptr[d] + pos] = s;
}

// ---------------------------------------------------------------------------
// agg[b,d,:] = sum over incoming edges of m[b,src,:]  (atomic-free gather)
// block = dst node, 256 threads: (b = tid>>5, bf16 pair h2 = tid&31)
// ---------------------------------------------------------------------------
__global__ __launch_bounds__(256) void agg_kernel(const int* __restrict__ rowptr,
                                                  const int* __restrict__ srcsort,
                                                  const __bf16* __restrict__ m_bf,
                                                  __bf16* __restrict__ agg_bf) {
  const int d   = blockIdx.x;
  const int tid = threadIdx.x;
  const int b   = tid >> 5;
  const int h2  = tid & 31;
  const int beg = rowptr[d], end = rowptr[d + 1];
  const bf16x2* mv = (const bf16x2*)m_bf;

  float a0 = 0.0f, a1 = 0.0f;
  for (int j = beg; j < end; ++j) {
    const int s = srcsort[j];
    const bf16x2 v = mv[((size_t)b * Nc + s) * 32 + h2];
    a0 += (float)v[0];
    a1 += (float)v[1];
  }
  bf16x2 o;
  o[0] = (__bf16)a0;
  o[1] = (__bf16)a1;
  ((bf16x2*)agg_bf)[((size_t)b * Nc + d) * 32 + h2] = o;
}

// ---------------------------------------------------------------------------
// m2 = concat(agg, x_t) @ W_mix^T + b_mix   [B*N,128]x[128,64]
// ---------------------------------------------------------------------------
__global__ __launch_bounds__(128) void mix_kernel(const __bf16* __restrict__ agg_bf,
                                                  const float* __restrict__ x,
                                                  const __bf16* __restrict__ WmixB,
                                                  const float* __restrict__ bmix,
                                                  __bf16* __restrict__ m2_bf, int t) {
  const int lane = threadIdx.x & 31;
  const int wave = threadIdx.x >> 5;
  const int row0 = blockIdx.x * 16;
  const int g0   = wave * 16;
  const int mrow = lane & 15;
  const int kb   = (lane >> 4) * 8;
  const int col  = g0 + (lane & 15);

  const int row_g = row0 + mrow;
  const int bidx  = row_g >> 11;             // N = 2048
  const int nidx  = row_g & (Nc - 1);
  const __bf16* aggp = agg_bf + (size_t)row_g * Hc;
  const float*  xp   = x + (((size_t)bidx * Tc + t) * Nc + nidx) * Fc;

  v8f acc;
  const float bias = bmix[col];
#pragma unroll
  for (int v = 0; v < 8; ++v) acc[v] = bias;

#pragma unroll
  for (int k0 = 0; k0 < Hc + Fc; k0 += 32) {
    v16bf a = (k0 < Hc) ? frag_load(aggp + k0 + kb)
                        : frag_load(xp + (k0 - Hc) + kb);
    v16bf b = frag_load(WmixB + (size_t)col * (Hc + Fc) + k0 + kb);
    acc = WMMA_BF16(a, b, acc);
  }

  const int rbase = row0 + 8 * (lane >> 4);
#pragma unroll
  for (int v = 0; v < 8; ++v)
    m2_bf[(size_t)(rbase + v) * Hc + col] = (__bf16)acc[v];
}

// ---------------------------------------------------------------------------
// Per-node GRU + readout + masked loss. block = node, 384 thr = 12 waves.
// ---------------------------------------------------------------------------
template <typename TW>
__global__ __launch_bounds__(384) void gru_kernel(
    const __bf16* __restrict__ m2_bf, float* __restrict__ state,
    __bf16* __restrict__ state_bf,
    const TW* __restrict__ Wih, const TW* __restrict__ Whh,
    const float* __restrict__ bih, const float* __restrict__ bhh,
    const float* __restrict__ Wread, const float* __restrict__ bread,
    const float* __restrict__ targets, const unsigned char* __restrict__ tmask,
    float* __restrict__ loss_accum, int t) {
  __shared__ float lds_gi[Bc][Gc];
  __shared__ float lds_gh[Bc][Gc];
  __shared__ float lds_red[Bc][Hc];

  const int n    = blockIdx.x;
  const int lane = threadIdx.x & 31;
  const int wave = threadIdx.x >> 5;       // 0..11
  const int g0   = wave * 16;
  const int mrow = lane & 15;              // batch row (valid < 8)
  const int kb   = (lane >> 4) * 8;
  const int gcol = g0 + (lane & 15);

  v8f ci, ch;
  const float bi_ = bih[(size_t)n * Gc + gcol];
  const float bh_ = bhh[(size_t)n * Gc + gcol];
#pragma unroll
  for (int v = 0; v < 8; ++v) { ci[v] = bi_; ch[v] = bh_; }

  const __bf16* m2p = m2_bf    + ((size_t)mrow * Nc + n) * Hc;  // valid iff mrow<8
  const __bf16* stp = state_bf + ((size_t)mrow * Nc + n) * Hc;
  const TW*     wip = Wih + ((size_t)n * Gc + gcol) * Hc;
  const TW*     whp = Whh + ((size_t)n * Gc + gcol) * Hc;
  __builtin_prefetch(wip, 0, 1);           // global_prefetch_b8
  __builtin_prefetch(whp, 0, 1);

#pragma unroll
  for (int k0 = 0; k0 < Hc; k0 += 32) {
    const v16bf ai = (mrow < Bc) ? frag_load(m2p + k0 + kb) : frag_zero();
    const v16bf ah = (mrow < Bc) ? frag_load(stp + k0 + kb) : frag_zero();
    const v16bf wbi = frag_load(wip + k0 + kb);
    const v16bf wbh = frag_load(whp + k0 + kb);
    ci = WMMA_BF16(ai, wbi, ci);
    ch = WMMA_BF16(ah, wbh, ch);
  }

  // D rows v + 8*(lane>>4): lanes 0-15 hold the valid batch rows 0..7
  if ((lane >> 4) == 0) {
#pragma unroll
    for (int v = 0; v < 8; ++v) {
      lds_gi[v][gcol] = ci[v];
      lds_gh[v][gcol] = ch[v];
    }
  }
  __syncthreads();

  const int tid = threadIdx.x;
  if (tid < Bc * Hc) {
    const int b = tid >> 6, h = tid & 63;
    const float ir = lds_gi[b][h],           hr = lds_gh[b][h];
    const float iz = lds_gi[b][Hc + h],      hz = lds_gh[b][Hc + h];
    const float inn = lds_gi[b][2 * Hc + h], hn = lds_gh[b][2 * Hc + h];
    const float r  = 1.0f / (1.0f + __expf(-(ir + hr)));
    const float z  = 1.0f / (1.0f + __expf(-(iz + hz)));
    const float nc = tanhf(inn + r * hn);
    const size_t sidx = ((size_t)b * Nc + n) * Hc + h;
    const float sold = state[sidx];
    const float snew = (1.0f - z) * nc + z * sold;
    state[sidx]    = snew;                 // in-place: this block owns node n
    state_bf[sidx] = (__bf16)snew;
    lds_red[b][h]  = snew * Wread[h];
  }
  __syncthreads();

  if (tid < Bc) {
    float s = bread[0];
#pragma unroll 8
    for (int h = 0; h < Hc; ++h) s += lds_red[tid][h];
    const size_t oidx = ((size_t)tid * Tc + t) * Nc + n;
    const float d = s - targets[oidx];
    if (tmask[oidx]) atomicAdd(loss_accum, 0.5f * d * d);
  }
}

// ---------------------------------------------------------------------------
// helpers
// ---------------------------------------------------------------------------
__global__ __launch_bounds__(256) void masksum_kernel(const unsigned char* __restrict__ mask,
                                                      float* __restrict__ acc, int total) {
  __shared__ float red[256];
  float v = 0.0f;
  for (int i = blockIdx.x * 256 + threadIdx.x; i < total; i += gridDim.x * 256)
    v += mask[i] ? 1.0f : 0.0f;
  red[threadIdx.x] = v;
  __syncthreads();
  for (int s = 128; s > 0; s >>= 1) {
    if (threadIdx.x < s) red[threadIdx.x] += red[threadIdx.x + s];
    __syncthreads();
  }
  if (threadIdx.x == 0) atomicAdd(acc, red[0]);
}

__global__ __launch_bounds__(256) void convert_bf16_kernel(const float* __restrict__ src,
                                                           __bf16* __restrict__ dst,
                                                           size_t count) {
  const size_t i = (size_t)blockIdx.x * 256 + threadIdx.x;
  if (i < count) dst[i] = (__bf16)src[i];
}

__global__ void finalize_kernel(const float* __restrict__ acc, float* __restrict__ out) {
  out[0] = acc[0] / acc[1];
}

// ---------------------------------------------------------------------------
extern "C" void kernel_launch(void* const* d_in, const int* in_sizes, int n_in,
                              void* d_out, int out_size, void* d_ws, size_t ws_size,
                              hipStream_t stream) {
  const float*         x       = (const float*)d_in[0];
  const float*         targets = (const float*)d_in[1];
  const unsigned char* tmask   = (const unsigned char*)d_in[2];
  const int*           edges   = (const int*)d_in[3];
  const float* Wmsg  = (const float*)d_in[4];
  const float* bmsg  = (const float*)d_in[5];
  const float* Wmix  = (const float*)d_in[6];
  const float* bmix  = (const float*)d_in[7];
  const float* Wih   = (const float*)d_in[8];
  const float* Whh   = (const float*)d_in[9];
  const float* bih   = (const float*)d_in[10];
  const float* bhh   = (const float*)d_in[11];
  const float* Wread = (const float*)d_in[12];
  const float* bread = (const float*)d_in[13];
  float* out = (float*)d_out;

  // ---- workspace carve-up (256B aligned chunks) ----
  char* p = (char*)d_ws;
  auto alloc = [&](size_t bytes) -> char* {
    char* r = p;
    p += (bytes + 255) & ~(size_t)255;
    return r;
  };
  float*  state    = (float*)alloc((size_t)BNH * sizeof(float));
  float*  accum    = (float*)alloc(2 * sizeof(float));     // [0]=loss, [1]=mask
  __bf16* state_bf = (__bf16*)alloc((size_t)BNH * sizeof(__bf16));
  __bf16* m_bf     = (__bf16*)alloc((size_t)BNH * sizeof(__bf16));
  __bf16* agg_bf   = (__bf16*)alloc((size_t)BNH * sizeof(__bf16));
  __bf16* m2_bf    = (__bf16*)alloc((size_t)BNH * sizeof(__bf16));
  int*    rowptr   = (int*)alloc((Nc + 1) * sizeof(int));
  int*    counts   = (int*)alloc(Nc * sizeof(int));
  int*    srcsort  = (int*)alloc(Ec * sizeof(int));
  __bf16* WmsgB    = (__bf16*)alloc((size_t)Hc * Hc * sizeof(__bf16));
  __bf16* WmixB    = (__bf16*)alloc((size_t)Hc * (Hc + Fc) * sizeof(__bf16));
  const size_t wcount = (size_t)Nc * Gc * Hc;              // 25,165,824
  __bf16* WihB     = (__bf16*)alloc(wcount * sizeof(__bf16));
  __bf16* WhhB     = (__bf16*)alloc(wcount * sizeof(__bf16));
  const bool use_bf16_w = ((size_t)(p - (char*)d_ws) <= ws_size);

  // ---- one-time init ----
  hipMemsetAsync(state, 0, (size_t)BNH * sizeof(float), stream);
  hipMemsetAsync(state_bf, 0, (size_t)BNH * sizeof(__bf16), stream);
  hipMemsetAsync(accum, 0, 2 * sizeof(float), stream);
  hipMemsetAsync(counts, 0, Nc * sizeof(int), stream);
  masksum_kernel<<<2048, 256, 0, stream>>>(tmask, accum + 1, Bc * Tc * Nc);

  // small weights -> bf16 (always; part of base footprint)
  convert_bf16_kernel<<<(Hc * Hc + 255) / 256, 256, 0, stream>>>(Wmsg, WmsgB, Hc * Hc);
  convert_bf16_kernel<<<(Hc * (Hc + Fc) + 255) / 256, 256, 0, stream>>>(Wmix, WmixB,
                                                                        Hc * (Hc + Fc));
  // big per-node weights -> bf16 (L2-resident afterwards)
  if (use_bf16_w) {
    const int cblocks = (int)((wcount + 255) / 256);
    convert_bf16_kernel<<<cblocks, 256, 0, stream>>>(Wih, WihB, wcount);
    convert_bf16_kernel<<<cblocks, 256, 0, stream>>>(Whh, WhhB, wcount);
  }

  // CSR-by-destination (atomic-free per-step aggregation)
  count_kernel<<<(Ec + 255) / 256, 256, 0, stream>>>(edges, counts);
  scan_kernel<<<1, 1, 0, stream>>>(counts, rowptr);
  fill_kernel<<<(Ec + 255) / 256, 256, 0, stream>>>(edges, rowptr, counts, srcsort);

  // ---- sequential scan over T ----
  for (int t = 0; t < Tc; ++t) {
    msg_kernel<<<BN / 16, 128, 0, stream>>>(state_bf, WmsgB, bmsg, m_bf);
    agg_kernel<<<Nc, 256, 0, stream>>>(rowptr, srcsort, m_bf, agg_bf);
    mix_kernel<<<BN / 16, 128, 0, stream>>>(agg_bf, x, WmixB, bmix, m2_bf, t);
    if (use_bf16_w)
      gru_kernel<__bf16><<<Nc, 384, 0, stream>>>(m2_bf, state, state_bf, WihB, WhhB,
                                                 bih, bhh, Wread, bread, targets,
                                                 tmask, accum, t);
    else
      gru_kernel<float><<<Nc, 384, 0, stream>>>(m2_bf, state, state_bf, Wih, Whh,
                                                bih, bhh, Wread, bread, targets,
                                                tmask, accum, t);
  }

  finalize_kernel<<<1, 1, 0, stream>>>(accum, out);
}
